// TemporalGNN_1975684956786
// MI455X (gfx1250) — matrix operand order repaired and checked by reading, not compiled, not run on hardware
//
#include <hip/hip_runtime.h>
#include <hip/hip_bf16.h>
#include <math.h>

#define NNODES 50000
#define TSTEPS 8
#define NEDGES 800000
#define FIN 9
#define HDIM 64

typedef __attribute__((ext_vector_type(16))) _Float16 v16h;
typedef __attribute__((ext_vector_type(8)))  float    v8f;

// ---------------- CDNA5 async copy to LDS (ASYNCcnt-tracked) ---------------

__device__ __forceinline__ void async_copy_b128(unsigned lds_off, const void* gptr) {
  asm volatile("global_load_async_to_lds_b128 %0, %1, off"
               :: "v"(lds_off), "v"((unsigned long long)(uintptr_t)gptr)
               : "memory");
}
__device__ __forceinline__ void wait_async0() {
  asm volatile("s_wait_asynccnt 0x0" ::: "memory");
}
// flat pointers to LDS are {aperture, 32'offset}: truncation yields LDS offset
__device__ __forceinline__ unsigned lds_offset_of(const void* p) {
  return (unsigned)(uintptr_t)p;
}

// ---------------- WMMA fragment helpers (CDNA5 16x16x32 f16 layouts) -------

// A fragment from an LDS-resident f32 tile [16][64]: one row per lane
// (lanes 0-15: K 0..7 & 16..23, lanes 16-31: K 8..15 & 24..31), cvt on load.
__device__ __forceinline__ v16h load_a_frag_lds(const float* sA, int k_base, int lane) {
  int r  = lane & 15;
  int k0 = k_base + ((lane & 16) ? 8 : 0);
  const float* p = sA + r * HDIM + k0;
  v16h a;
#pragma unroll
  for (int j = 0; j < 8; ++j) a[j] = (_Float16)p[j];
#pragma unroll
  for (int j = 0; j < 8; ++j) a[8 + j] = (_Float16)p[16 + j];
  return a;
}

// ---------------- small utility kernels ------------------------------------

__global__ void zero_f32_k(float* __restrict__ p, int n) {
  int i = blockIdx.x * blockDim.x + threadIdx.x;
  if (i < n) p[i] = 0.0f;
}

__global__ void set_const_k(float* __restrict__ p, int n, float v) {
  int i = blockIdx.x * blockDim.x + threadIdx.x;
  if (i < n) p[i] = v;
}

__global__ void count_deg_k(const int* __restrict__ dst, float* __restrict__ deg) {
  int e = blockIdx.x * blockDim.x + threadIdx.x;
  if (e < NEDGES) atomicAdd(&deg[dst[e]], 1.0f);
}

__global__ void rsqrt_k(const float* __restrict__ deg, float* __restrict__ dinv) {
  int i = blockIdx.x * blockDim.x + threadIdx.x;
  if (i < NNODES) dinv[i] = rsqrtf(deg[i]);
}

// h1 = x @ W1   (x: [N,9], W1: [9,64]) -- K=9, plain VALU is fine.
__global__ void xw1_k(const float* __restrict__ x, const float* __restrict__ W1,
                      float* __restrict__ out) {
  int t = blockIdx.x * blockDim.x + threadIdx.x;
  if (t >= NNODES * HDIM) return;
  int n = t >> 6, j = t & 63;
  const float* xr = x + (size_t)n * FIN;
  float s = 0.0f;
#pragma unroll
  for (int k = 0; k < FIN; ++k) s = fmaf(xr[k], W1[k * HDIM + j], s);
  out[t] = s;
}

// scatter: agg[dst] += h[src] * dinv[src]*dinv[dst]; 16 threads/edge, 4 feats each
__global__ void edge_agg_k(const int* __restrict__ src, const int* __restrict__ dst,
                           const float* __restrict__ dinv, const float* __restrict__ h,
                           float* __restrict__ agg) {
  int t = blockIdx.x * blockDim.x + threadIdx.x;
  int e = t >> 4;
  if (e >= NEDGES) return;
  int f4 = (t & 15) << 2;
  int s = src[e], d = dst[e];
  float coef = dinv[s] * dinv[d];
  const float4 hv = *reinterpret_cast<const float4*>(h + (size_t)s * HDIM + f4);
  float* ap = agg + (size_t)d * HDIM + f4;
  atomicAdd(ap + 0, hv.x * coef);
  atomicAdd(ap + 1, hv.y * coef);
  atomicAdd(ap + 2, hv.z * coef);
  atomicAdd(ap + 3, hv.w * coef);
}

// out = relu(agg + h * dinv^2 + b)
__global__ void self_bias_relu_k(const float* __restrict__ agg, const float* __restrict__ h,
                                 const float* __restrict__ dinv, const float* __restrict__ b,
                                 float* __restrict__ out) {
  int t = blockIdx.x * blockDim.x + threadIdx.x;
  if (t >= NNODES * HDIM) return;
  int n = t >> 6, j = t & 63;
  float di = dinv[n];
  float v = agg[t] + h[t] * di * di + b[j];
  out[t] = v > 0.0f ? v : 0.0f;
}

// Pre-pack a [64 x ncols] B matrix (optionally stored transposed as
// [ncols x 64] like w_ih/w_hh) into WMMA fragment order:
//   dst[((ct*2+kbi)*32 + lane)*16 + j]  with c = ct*16+(lane&15),
//   k = kbi*32 + ((lane&16)?8:0) + (j&7) + ((j&8)?16:0)
// so each lane's 16-half fragment is one contiguous 32B chunk.
__global__ void pack_b_frags_k(const float* __restrict__ src, _Float16* __restrict__ dst,
                               int ncols, int transpose) {
  int t = blockIdx.x * blockDim.x + threadIdx.x;
  int total = (ncols >> 4) * 2 * 32 * 16;
  if (t >= total) return;
  int j   = t & 15;
  int l   = (t >> 4) & 31;
  int fi  = t >> 9;
  int ct  = fi >> 1, kbi = fi & 1;
  int c   = ct * 16 + (l & 15);
  int k   = kbi * 32 + ((l & 16) ? 8 : 0) + (j & 7) + ((j & 8) ? 16 : 0);
  float v = transpose ? src[c * HDIM + k] : src[k * ncols + c];
  dst[t] = (_Float16)v;
}

// ---------------- WMMA GEMM: C[N,64] = A[N,64] @ B[64,64] ------------------
// one 16-row tile per block; async-stage the 4KB A tile into LDS once, then
// 4 waves each compute one 16-col tile; K=64 -> 2 WMMAs/wave. B fragments are
// pre-packed (2 x global_load_b128 per fragment).
__global__ __launch_bounds__(128) void gemm_h64_wmma_k(const float* __restrict__ A,
                                                       const _Float16* __restrict__ Bpk,
                                                       float* __restrict__ C) {
  __shared__ float at[16 * HDIM];
  int tid  = threadIdx.x;
  int lane = tid & 31;
  int wave = tid >> 5;
  int row_base = blockIdx.x * 16;

  const char* ag  = (const char*)(A + (size_t)row_base * HDIM);
  unsigned abase  = lds_offset_of(at);
  async_copy_b128(abase + tid * 16u,         ag + tid * 16);
  async_copy_b128(abase + 2048u + tid * 16u, ag + 2048 + tid * 16);
  wait_async0();
  __syncthreads();

  v16h a0 = load_a_frag_lds(at, 0,  lane);
  v16h a1 = load_a_frag_lds(at, 32, lane);
  const v16h* BF = (const v16h*)Bpk;
  v16h b0 = BF[(size_t)(wave * 2 + 0) * 32 + lane];
  v16h b1 = BF[(size_t)(wave * 2 + 1) * 32 + lane];
  v8f acc = {};
  acc = __builtin_amdgcn_wmma_f32_16x16x32_f16(false, a0, false, b0, (short)0, acc, false, false);
  acc = __builtin_amdgcn_wmma_f32_16x16x32_f16(false, a1, false, b1, (short)0, acc, false, false);

  int c  = wave * 16 + (lane & 15);
  int r0 = row_base + ((lane & 16) ? 8 : 0);
#pragma unroll
  for (int r = 0; r < 8; ++r) C[(size_t)(r0 + r) * HDIM + c] = acc[r];
}

// ---------------- fused LSTM step ------------------------------------------
// gates = X @ WihT + Hin @ WhhT + bias; X/Hin 16-row tiles async-staged to
// LDS, shared by all 4 waves; wave w computes gate w's 64 cols (4 col tiles
// x 4 WMMAs). Pointwise update: c in place, h double buffered.
__global__ __launch_bounds__(128) void lstm_step_wmma_k(
    const float* __restrict__ X, const float* __restrict__ Hin,
    const _Float16* __restrict__ WihPk, const _Float16* __restrict__ WhhPk,
    const float* __restrict__ b_ih, const float* __restrict__ b_hh,
    float* __restrict__ Cst, float* __restrict__ Hout) {
  __shared__ float xt[16 * HDIM];
  __shared__ float ht[16 * HDIM];
  __shared__ float g[16 * 256];
  int tid  = threadIdx.x;
  int lane = tid & 31;
  int wave = tid >> 5;
  int row_base = blockIdx.x * 16;

  const char* xg = (const char*)(X   + (size_t)row_base * HDIM);
  const char* hg = (const char*)(Hin + (size_t)row_base * HDIM);
  unsigned xbase = lds_offset_of(xt);
  unsigned hbase = lds_offset_of(ht);
  async_copy_b128(xbase + tid * 16u,         xg + tid * 16);
  async_copy_b128(xbase + 2048u + tid * 16u, xg + 2048 + tid * 16);
  async_copy_b128(hbase + tid * 16u,         hg + tid * 16);
  async_copy_b128(hbase + 2048u + tid * 16u, hg + 2048 + tid * 16);
  wait_async0();
  __syncthreads();

  v16h ax0 = load_a_frag_lds(xt, 0,  lane);
  v16h ax1 = load_a_frag_lds(xt, 32, lane);
  v16h ah0 = load_a_frag_lds(ht, 0,  lane);
  v16h ah1 = load_a_frag_lds(ht, 32, lane);

  const v16h* WIF = (const v16h*)WihPk;
  const v16h* WHF = (const v16h*)WhhPk;
#pragma unroll
  for (int ct = 0; ct < 4; ++ct) {
    int ct_g = wave * 4 + ct;               // global 16-col tile in [0,16)
    v8f acc = {};
    v16h b;
    b = WIF[(size_t)(ct_g * 2 + 0) * 32 + lane];
    acc = __builtin_amdgcn_wmma_f32_16x16x32_f16(false, ax0, false, b, (short)0, acc, false, false);
    b = WIF[(size_t)(ct_g * 2 + 1) * 32 + lane];
    acc = __builtin_amdgcn_wmma_f32_16x16x32_f16(false, ax1, false, b, (short)0, acc, false, false);
    b = WHF[(size_t)(ct_g * 2 + 0) * 32 + lane];
    acc = __builtin_amdgcn_wmma_f32_16x16x32_f16(false, ah0, false, b, (short)0, acc, false, false);
    b = WHF[(size_t)(ct_g * 2 + 1) * 32 + lane];
    acc = __builtin_amdgcn_wmma_f32_16x16x32_f16(false, ah1, false, b, (short)0, acc, false, false);
    int c   = ct_g * 16 + (lane & 15);
    int r0l = (lane & 16) ? 8 : 0;
    float bias = b_ih[c] + b_hh[c];
#pragma unroll
    for (int r = 0; r < 8; ++r) g[(r0l + r) * 256 + c] = acc[r] + bias;
  }
  __syncthreads();

  // pointwise: 16 rows x 64 hidden cols = 1024 elems, 8 per thread
#pragma unroll
  for (int it = 0; it < 8; ++it) {
    int idx = tid * 8 + it;
    int row = idx >> 6, hc = idx & 63;
    float gi = g[row * 256 + hc];
    float gf = g[row * 256 + 64 + hc];
    float gg = g[row * 256 + 128 + hc];
    float go = g[row * 256 + 192 + hc];
    float si = 1.0f / (1.0f + __expf(-gi));
    float sf = 1.0f / (1.0f + __expf(-gf));
    float so = 1.0f / (1.0f + __expf(-go));
    float tg = tanhf(gg);
    size_t gidx = (size_t)(row_base + row) * HDIM + hc;
    float cn = sf * Cst[gidx] + si * tg;
    Cst[gidx]  = cn;
    Hout[gidx] = so * tanhf(cn);
  }
}

__global__ void head_k(const float* __restrict__ H, const float* __restrict__ w,
                       const float* __restrict__ b, float* __restrict__ out) {
  int i = blockIdx.x * blockDim.x + threadIdx.x;
  if (i >= NNODES) return;
  const float* hr = H + (size_t)i * HDIM;
  float s = 0.0f;
#pragma unroll
  for (int k = 0; k < HDIM; ++k) s = fmaf(hr[k], w[k], s);
  out[i] = s + b[0];
}

// ---------------- host orchestration ---------------------------------------

extern "C" void kernel_launch(void* const* d_in, const int* in_sizes, int n_in,
                              void* d_out, int out_size, void* d_ws, size_t ws_size,
                              hipStream_t stream) {
  (void)in_sizes; (void)n_in; (void)out_size; (void)ws_size;

  const float* xs     = (const float*)d_in[0];   // [T,N,9]
  const int*   ei     = (const int*)  d_in[1];   // [T,2,E]
  const float* W1     = (const float*)d_in[2];
  const float* b1     = (const float*)d_in[3];
  const float* W2     = (const float*)d_in[4];
  const float* b2     = (const float*)d_in[5];
  const float* w_ih   = (const float*)d_in[6];
  const float* w_hh   = (const float*)d_in[7];
  const float* b_ih   = (const float*)d_in[8];
  const float* b_hh   = (const float*)d_in[9];
  const float* head_w = (const float*)d_in[10];
  const float* head_b = (const float*)d_in[11];
  float* out = (float*)d_out;

  char* ws = (char*)d_ws;
  auto carve = [&](size_t bytes) -> void* {
    void* p = (void*)ws;
    ws += (bytes + 255) & ~(size_t)255;
    return p;
  };
  const size_t NH = (size_t)NNODES * HDIM;
  float* deg  = (float*)carve(NNODES * sizeof(float));
  float* dinv = (float*)carve(NNODES * sizeof(float));
  float* bufA = (float*)carve(NH * sizeof(float)); // h (pre-agg)
  float* bufB = (float*)carve(NH * sizeof(float)); // agg accumulator
  float* bufC = (float*)carve(NH * sizeof(float)); // post-relu / seq_t
  float* hA   = (float*)carve(NH * sizeof(float));
  float* hB   = (float*)carve(NH * sizeof(float));
  float* cS   = (float*)carve(NH * sizeof(float));
  _Float16* W2pk  = (_Float16*)carve(8  * 32 * 16 * sizeof(_Float16)); // 4 tiles x 2 kb
  _Float16* WihPk = (_Float16*)carve(32 * 32 * 16 * sizeof(_Float16)); // 16 tiles x 2 kb
  _Float16* WhhPk = (_Float16*)carve(32 * 32 * 16 * sizeof(_Float16));

  const int B256 = 256;
  const int gNH  = (int)((NH + B256 - 1) / B256);
  const int gN   = (NNODES + B256 - 1) / B256;
  const int gE   = (NEDGES + B256 - 1) / B256;
  const int gE16 = (int)(((size_t)NEDGES * 16 + B256 - 1) / B256);
  const int gTile = NNODES / 16; // 3125 exact

  // init LSTM state + pre-pack weights into WMMA fragment order
  zero_f32_k<<<gNH, B256, 0, stream>>>(hA, (int)NH);
  zero_f32_k<<<gNH, B256, 0, stream>>>(cS, (int)NH);
  pack_b_frags_k<<<(8  * 512 + B256 - 1) / B256, B256, 0, stream>>>(W2,   W2pk,  64,  0);
  pack_b_frags_k<<<(32 * 512 + B256 - 1) / B256, B256, 0, stream>>>(w_ih, WihPk, 256, 1);
  pack_b_frags_k<<<(32 * 512 + B256 - 1) / B256, B256, 0, stream>>>(w_hh, WhhPk, 256, 1);

  float* hcur = hA;
  float* hnxt = hB;
  for (int t = 0; t < TSTEPS; ++t) {
    const float* x_t = xs + (size_t)t * NNODES * FIN;
    const int* src = ei + (size_t)t * 2 * NEDGES;
    const int* dst = src + NEDGES;

    // symmetric norm coefficients
    set_const_k<<<gN, B256, 0, stream>>>(deg, NNODES, 1.0f);
    count_deg_k<<<gE, B256, 0, stream>>>(dst, deg);
    rsqrt_k<<<gN, B256, 0, stream>>>(deg, dinv);

    // GCN layer 1
    xw1_k<<<gNH, B256, 0, stream>>>(x_t, W1, bufA);
    zero_f32_k<<<gNH, B256, 0, stream>>>(bufB, (int)NH);
    edge_agg_k<<<gE16, B256, 0, stream>>>(src, dst, dinv, bufA, bufB);
    self_bias_relu_k<<<gNH, B256, 0, stream>>>(bufB, bufA, dinv, b1, bufC);

    // GCN layer 2 (WMMA GEMM)
    gemm_h64_wmma_k<<<gTile, 128, 0, stream>>>(bufC, W2pk, bufA);
    zero_f32_k<<<gNH, B256, 0, stream>>>(bufB, (int)NH);
    edge_agg_k<<<gE16, B256, 0, stream>>>(src, dst, dinv, bufA, bufB);
    self_bias_relu_k<<<gNH, B256, 0, stream>>>(bufB, bufA, dinv, b2, bufC);

    // LSTM step (WMMA gates + fused pointwise)
    lstm_step_wmma_k<<<gTile, 128, 0, stream>>>(bufC, hcur, WihPk, WhhPk,
                                                b_ih, b_hh, cS, hnxt);
    float* tmp = hcur; hcur = hnxt; hnxt = tmp;
  }

  head_k<<<gN, B256, 0, stream>>>(hcur, head_w, head_b, out);
}
